// LongGemini_33792802685373
// MI455X (gfx1250) — matrix operand order repaired and checked by reading, not compiled, not run on hardware
//
#include <hip/hip_runtime.h>
#include <cstdint>
#include <cstddef>

// ---------------- problem constants ----------------
#define S_LEN    2048
#define D_MODEL  2048
#define NH       16
#define NKH      2
#define DHEAD    128
#define NLAYER   4
#define VOCAB    10000
#define FF_INNER 8192

// ---------------- WMMA types ----------------
typedef __attribute__((ext_vector_type(16))) __bf16 v16bf;
typedef __attribute__((ext_vector_type(8)))  float  v8f;

struct __attribute__((aligned(16))) U4 { unsigned int x, y, z, w; };
union FragAB { v16bf v; U4 u[2]; };
union Pack8  { U4 u; __bf16 e[8]; };

static __device__ __forceinline__ v8f wmma_bf16(v16bf a, v16bf b, v8f c) {
  // D = A(16x32 bf16) * B(32x16 bf16) + C(16x16 f32)
  return __builtin_amdgcn_wmma_f32_16x16x32_bf16(false, a, false, b, (short)0, c,
                                                 false, false);
}

static __device__ __forceinline__ float gelu_tanh(float x) {
  float x3 = x * x * x;
  return 0.5f * x * (1.0f + tanhf(0.7978845608028654f * (x + 0.044715f * x3)));
}

// ---------------- elementwise / prep kernels ----------------

__global__ __launch_bounds__(256)
void gather_kernel(const int* __restrict__ text, const float* __restrict__ emb,
                   float* __restrict__ out, int D) {
  int s = blockIdx.y;
  int d = blockIdx.x * 256 + threadIdx.x;
  out[(size_t)s * D + d] = emb[(size_t)text[s] * D + d];
}

__global__ __launch_bounds__(256)
void ln_kernel(const float* __restrict__ in, const float* __restrict__ g,
               const float* __restrict__ b, float* __restrict__ o32,
               __bf16* __restrict__ o16, int D) {
  int row = blockIdx.x;
  const float* x = in + (size_t)row * D;
  float s = 0.f, s2 = 0.f;
  for (int i = threadIdx.x; i < D; i += 256) {
    float v = x[i]; s += v; s2 += v * v;
  }
  for (int off = 16; off > 0; off >>= 1) {
    s  += __shfl_xor(s,  off, 32);
    s2 += __shfl_xor(s2, off, 32);
  }
  __shared__ float rs[8], rs2[8];
  int wid = threadIdx.x >> 5, lane = threadIdx.x & 31;
  if (lane == 0) { rs[wid] = s; rs2[wid] = s2; }
  __syncthreads();
  s = 0.f; s2 = 0.f;
  for (int i = 0; i < 8; i++) { s += rs[i]; s2 += rs2[i]; }
  float mean = s / (float)D;
  float var  = s2 / (float)D - mean * mean;
  float inv  = rsqrtf(var + 1e-5f);
  for (int i = threadIdx.x; i < D; i += 256) {
    float v = (x[i] - mean) * inv * g[i] + b[i];
    size_t idx = (size_t)row * D + i;
    if (o32) o32[idx] = v;
    if (o16) o16[idx] = (__bf16)v;
  }
}

// LayerNorm over DH=128 with learnable scale, zero bias (qk-norm)
__global__ __launch_bounds__(256)
void qknorm_kernel(const float* __restrict__ in, const float* __restrict__ g,
                   __bf16* __restrict__ out, int rows) {
  int r = blockIdx.x * 8 + (threadIdx.x >> 5);
  int lane = threadIdx.x & 31;
  if (r >= rows) return;
  const float* x = in + (size_t)r * DHEAD;
  float vals[4];
  float s = 0.f, s2 = 0.f;
#pragma unroll
  for (int j = 0; j < 4; j++) {
    vals[j] = x[lane + 32 * j]; s += vals[j]; s2 += vals[j] * vals[j];
  }
  for (int off = 16; off > 0; off >>= 1) {
    s  += __shfl_xor(s,  off, 32);
    s2 += __shfl_xor(s2, off, 32);
  }
  float mean = s * (1.0f / DHEAD);
  float var  = s2 * (1.0f / DHEAD) - mean * mean;
  float inv  = rsqrtf(var + 1e-5f);
#pragma unroll
  for (int j = 0; j < 4; j++) {
    int c = lane + 32 * j;
    out[(size_t)r * DHEAD + c] = (__bf16)((vals[j] - mean) * inv * g[c]);
  }
}

__global__ __launch_bounds__(256)
void cast_kernel(const float* __restrict__ in, __bf16* __restrict__ out, int n) {
  int i = blockIdx.x * 256 + threadIdx.x;
  if (i < n) out[i] = (__bf16)in[i];
}

// in: f32 [R][C]  ->  out: bf16 [C][R]  (weight transpose + downcast)
__global__ __launch_bounds__(256)
void tc_kernel(const float* __restrict__ in, __bf16* __restrict__ out, int R, int C) {
  __shared__ float t[32][33];
  int c0 = blockIdx.x * 32, r0 = blockIdx.y * 32;
  for (int i = threadIdx.x; i < 1024; i += 256) {
    int tr = i >> 5, tc = i & 31;
    int r = r0 + tr, c = c0 + tc;
    t[tr][tc] = (r < R && c < C) ? in[(size_t)r * C + c] : 0.f;
  }
  __syncthreads();
  for (int i = threadIdx.x; i < 1024; i += 256) {
    int tr = i >> 5, tc = i & 31;
    int oc = c0 + tr, orow = r0 + tc;
    if (oc < C && orow < R)
      out[(size_t)oc * R + orow] = (__bf16)t[tc][tr];
  }
}

// ---------------- tiled bf16 WMMA GEMM (double-buffered LDS) ----------------
// C[M,N] = A[M,K](bf16,row) * Bt[N,K](bf16,row, i.e. B column-major)
// EPI 0: out16 = bf16(acc + bias?)      (bias may be null)
// EPI 1: out16 = bf16(gelu(acc + bias))
// EPI 2: out32 = acc + bias? + res?
template <int EPI>
__global__ __launch_bounds__(256)
void gemm_bf16(const __bf16* __restrict__ A, const __bf16* __restrict__ Bt,
               float* __restrict__ out32, __bf16* __restrict__ out16,
               const float* __restrict__ bias, const float* __restrict__ res,
               int M, int N, int K) {
  __shared__ __bf16 As[2][128 * 32];
  __shared__ __bf16 Bs[2][128 * 32];
  const int tid  = threadIdx.x;
  const int lane = tid & 31, wid = tid >> 5;
  const int half = lane >> 4, ln = lane & 15;
  const int wm = wid & 1, wn = wid >> 1;        // 2 x 4 waves -> wave tile 64x32
  const int r0 = blockIdx.y * 128, c0 = blockIdx.x * 128;

  v8f acc[4][2];
  v8f zero = {0.f, 0.f, 0.f, 0.f, 0.f, 0.f, 0.f, 0.f};
#pragma unroll
  for (int i = 0; i < 4; i++)
#pragma unroll
    for (int j = 0; j < 2; j++) acc[i][j] = zero;

  const int lrow  = tid >> 1;   // 0..127
  const int lhalf = tid & 1;    // which 16-element half of the 32-wide k slab
  const int brow  = c0 + lrow;
  const __bf16* arow_p = A + (size_t)(r0 + lrow) * K + lhalf * 16;
  const __bf16* brow_p = (brow < N) ? (Bt + (size_t)brow * K + lhalf * 16) : nullptr;

  U4 ra0, ra1, rb0, rb1;
  auto load_tile = [&](int k0) {
    const __bf16* ap = arow_p + k0;
    ra0 = *(const U4*)ap; ra1 = *(const U4*)(ap + 8);
    U4 z = {0, 0, 0, 0}; rb0 = z; rb1 = z;
    if (brow_p) {
      const __bf16* bp = brow_p + k0;
      rb0 = *(const U4*)bp; rb1 = *(const U4*)(bp + 8);
    }
  };
  auto store_tile = [&](int buf) {
    *(U4*)&As[buf][lrow * 32 + lhalf * 16]     = ra0;
    *(U4*)&As[buf][lrow * 32 + lhalf * 16 + 8] = ra1;
    *(U4*)&Bs[buf][lrow * 32 + lhalf * 16]     = rb0;
    *(U4*)&Bs[buf][lrow * 32 + lhalf * 16 + 8] = rb1;
  };

  const int nsteps = K >> 5;
  load_tile(0);
  store_tile(0);
  __syncthreads();

  for (int s = 0; s < nsteps; s++) {
    const int cur = s & 1;
    const bool more = (s + 1) < nsteps;
    if (more) load_tile((s + 1) << 5);           // global loads for next slab, issued early
    if (s + 2 < nsteps) {                        // gfx1250 global_prefetch_b8, two slabs ahead
      __builtin_prefetch(arow_p + ((s + 2) << 5), 0, 1);
      if (brow_p) __builtin_prefetch(brow_p + ((s + 2) << 5), 0, 1);
    }

    FragAB bfrag[2];
#pragma unroll
    for (int tj = 0; tj < 2; tj++) {
      const __bf16* p = &Bs[cur][(wn * 32 + tj * 16 + ln) * 32 + half * 16];
      bfrag[tj].u[0] = *(const U4*)p;
      bfrag[tj].u[1] = *(const U4*)(p + 8);
    }
#pragma unroll
    for (int ti = 0; ti < 4; ti++) {
      FragAB afrag;
      const __bf16* p = &As[cur][(wm * 64 + ti * 16 + ln) * 32];
      afrag.u[0] = *(const U4*)(p + 16 * half);
      afrag.u[1] = *(const U4*)(p + 16 * half + 8);
#pragma unroll
      for (int tj = 0; tj < 2; tj++)
        acc[ti][tj] = wmma_bf16(afrag.v, bfrag[tj].v, acc[ti][tj]);
    }

    if (more) store_tile(cur ^ 1);               // write ping-pong buffer
    __syncthreads();                             // single barrier per k-step
  }

#pragma unroll
  for (int ti = 0; ti < 4; ti++) {
#pragma unroll
    for (int tj = 0; tj < 2; tj++) {
      int row0 = r0 + wm * 64 + ti * 16 + half * 8;
      int col  = c0 + wn * 32 + tj * 16 + ln;
      if (col < N) {
        float bv = bias ? bias[col] : 0.f;
#pragma unroll
        for (int v = 0; v < 8; v++) {
          int row = row0 + v;
          float x = acc[ti][tj][v] + bv;
          size_t idx = (size_t)row * N + col;
          if (EPI == 0)       out16[idx] = (__bf16)x;
          else if (EPI == 1)  out16[idx] = (__bf16)gelu_tanh(x);
          else {
            if (res) x += res[idx];
            out32[idx] = x;
          }
        }
      }
    }
  }
}

// ---------------- flash attention (one wave per 16-query tile per head) ----------------
__global__ __launch_bounds__(32)
void attn_kernel(const __bf16* __restrict__ Q, const __bf16* __restrict__ Kk,
                 const __bf16* __restrict__ V, __bf16* __restrict__ O) {
  __shared__ __bf16 Vt[128 * 32];   // V^T tile: [dh=128][key=32]
  __shared__ __bf16 Ps[16 * 32];    // probability tile [16 q][32 key]

  const int lane = threadIdx.x;
  const int half = lane >> 4, ln = lane & 15;
  const int q0   = blockIdx.x * 16;
  const int h    = blockIdx.y;
  const int kvh  = h >> 3;                 // GQA: 16 q-heads -> 2 kv-heads
  const int QLD  = NH * DHEAD;             // 2048
  const int KLD  = NKH * DHEAD;            // 256
  const float scale = 0.08838834764831845f; // 1/sqrt(128)

  // preload Q fragments (4 k-steps of 32 over DH=128)
  FragAB qf[4];
  const __bf16* qbase = Q + (size_t)(q0 + ln) * QLD + h * DHEAD;
#pragma unroll
  for (int j = 0; j < 4; j++) {
    qf[j].u[0] = *(const U4*)(qbase + j * 32 + 16 * half);
    qf[j].u[1] = *(const U4*)(qbase + j * 32 + 16 * half + 8);
  }

  v8f zero = {0.f, 0.f, 0.f, 0.f, 0.f, 0.f, 0.f, 0.f};
  v8f o[8];
  float rm[8], rl[8];
#pragma unroll
  for (int t = 0; t < 8; t++) o[t] = zero;
#pragma unroll
  for (int v = 0; v < 8; v++) { rm[v] = -3.0e38f; rl[v] = 0.f; }

  for (int kb = 0; kb <= q0 + 15; kb += 32) {
    // stage V^T (each lane owns key row kb+lane)
    const __bf16* vrow = V + (size_t)(kb + lane) * KLD + kvh * DHEAD;
    for (int cc = 0; cc < 128; cc += 8) {
      Pack8 p; p.u = *(const U4*)(vrow + cc);
#pragma unroll
      for (int e = 0; e < 8; e++) Vt[(cc + e) * 32 + lane] = p.e[e];
    }
    __syncthreads();

    // scores: S = Q * K^T for 32 keys -> two 16x16 tiles
    v8f s0 = zero, s1 = zero;
#pragma unroll
    for (int j = 0; j < 4; j++) {
      FragAB kf0, kf1;
      const __bf16* kp0 = Kk + (size_t)(kb + ln) * KLD + kvh * DHEAD + j * 32 + 16 * half;
      kf0.u[0] = *(const U4*)kp0;        kf0.u[1] = *(const U4*)(kp0 + 8);
      const __bf16* kp1 = kp0 + (size_t)16 * KLD;
      kf1.u[0] = *(const U4*)kp1;        kf1.u[1] = *(const U4*)(kp1 + 8);
      s0 = wmma_bf16(qf[j].v, kf0.v, s0);
      s1 = wmma_bf16(qf[j].v, kf1.v, s1);
    }

    // online softmax per row (row = q0 + v + 8*half; cols spread across 16 lanes)
#pragma unroll
    for (int v = 0; v < 8; v++) {
      int row = q0 + v + 8 * half;
      int c0i = kb + ln, c1i = c0i + 16;
      float a = s0[v] * scale; if (c0i > row) a = -3.0e38f;
      float b = s1[v] * scale; if (c1i > row) b = -3.0e38f;
      float mx = fmaxf(a, b);
      for (int off = 8; off > 0; off >>= 1) mx = fmaxf(mx, __shfl_xor(mx, off, 16));
      float mnew  = fmaxf(rm[v], mx);
      float alpha = expf(rm[v] - mnew);
      float p0 = expf(a - mnew), p1 = expf(b - mnew);
      float sum = p0 + p1;
      for (int off = 8; off > 0; off >>= 1) sum += __shfl_xor(sum, off, 16);
      rl[v] = rl[v] * alpha + sum;
      rm[v] = mnew;
#pragma unroll
      for (int t = 0; t < 8; t++) o[t][v] *= alpha;
      Ps[(v + 8 * half) * 32 + ln]      = (__bf16)p0;
      Ps[(v + 8 * half) * 32 + 16 + ln] = (__bf16)p1;
    }
    __syncthreads();

    // P (16x32) as A fragment
    FragAB pf;
    const __bf16* pp = &Ps[ln * 32];
    pf.u[0] = *(const U4*)(pp + 16 * half);
    pf.u[1] = *(const U4*)(pp + 16 * half + 8);

    // O += P * V  (8 x 16-wide dh chunks)
#pragma unroll
    for (int t = 0; t < 8; t++) {
      FragAB vf;
      const __bf16* vp = &Vt[(t * 16 + ln) * 32 + 16 * half];
      vf.u[0] = *(const U4*)vp; vf.u[1] = *(const U4*)(vp + 8);
      o[t] = wmma_bf16(pf.v, vf.v, o[t]);
    }
    __syncthreads();
  }

#pragma unroll
  for (int v = 0; v < 8; v++) {
    float inv = 1.0f / rl[v];
    int row = q0 + v + 8 * half;
#pragma unroll
    for (int t = 0; t < 8; t++)
      O[(size_t)row * QLD + h * DHEAD + t * 16 + ln] = (__bf16)(o[t][v] * inv);
  }
}

// ---------------- host orchestration ----------------

static inline dim3 gemm_grid(int M, int N) { return dim3((N + 127) / 128, (M + 127) / 128); }
static inline dim3 tc_grid(int R, int C)   { return dim3((C + 31) / 32, (R + 31) / 32); }

extern "C" void kernel_launch(void* const* d_in, const int* in_sizes, int n_in,
                              void* d_out, int out_size, void* d_ws, size_t ws_size,
                              hipStream_t stream) {
  const int*   text = (const int*)d_in[0];
  const float* emb  = (const float*)d_in[1];
  const float* mng  = (const float*)d_in[2];
  const float* mnb  = (const float*)d_in[3];
  const float* bng  = (const float*)d_in[4];
  const float* bnb  = (const float*)d_in[5];
  const float* ang  = (const float*)d_in[6];
  const float* anb  = (const float*)d_in[7];
  const float* wq   = (const float*)d_in[8];
  const float* wk   = (const float*)d_in[9];
  const float* wv   = (const float*)d_in[10];
  const float* wo   = (const float*)d_in[11];
  const float* qng  = (const float*)d_in[12];
  const float* kng  = (const float*)d_in[13];
  const float* w1   = (const float*)d_in[14];
  const float* b1   = (const float*)d_in[15];
  const float* w2   = (const float*)d_in[16];
  const float* b2   = (const float*)d_in[17];
  const float* hng  = (const float*)d_in[18];
  const float* hnb  = (const float*)d_in[19];
  const float* hw   = (const float*)d_in[20];
  const float* hb   = (const float*)d_in[21];
  float* out = (float*)d_out;
  (void)in_sizes; (void)n_in; (void)out_size; (void)ws_size;

  char* ws = (char*)d_ws;
  size_t off = 0;
  auto alloc = [&](size_t bytes) -> void* {
    void* p = ws + off;
    off += (bytes + 255) & ~(size_t)255;
    return p;
  };

  __bf16* wT   = (__bf16*)alloc((size_t)VOCAB * D_MODEL * 2);        // reusable weight scratch
  float*  x32  = (float*) alloc((size_t)S_LEN * D_MODEL * 4);
  float*  h32  = (float*) alloc((size_t)S_LEN * D_MODEL * 4);
  float*  h232 = (float*) alloc((size_t)S_LEN * D_MODEL * 4);
  float*  q32  = (float*) alloc((size_t)S_LEN * NH * DHEAD * 4);
  float*  k32  = (float*) alloc((size_t)S_LEN * NKH * DHEAD * 4);
  float*  ff32 = (float*) alloc((size_t)S_LEN * D_MODEL * 4);
  __bf16* hn16 = (__bf16*)alloc((size_t)S_LEN * D_MODEL * 2);
  __bf16* q16  = (__bf16*)alloc((size_t)S_LEN * NH * DHEAD * 2);
  __bf16* k16  = (__bf16*)alloc((size_t)S_LEN * NKH * DHEAD * 2);
  __bf16* v16  = (__bf16*)alloc((size_t)S_LEN * NKH * DHEAD * 2);
  __bf16* a16  = (__bf16*)alloc((size_t)S_LEN * NH * DHEAD * 2);
  __bf16* h216 = (__bf16*)alloc((size_t)S_LEN * D_MODEL * 2);
  __bf16* g16  = (__bf16*)alloc((size_t)S_LEN * FF_INNER * 2);
  __bf16* xh16 = (__bf16*)alloc((size_t)S_LEN * D_MODEL * 2);

  const int HD = NH * DHEAD;    // 2048
  const int KD = NKH * DHEAD;   // 256

  // embed -> model LN -> x32
  gather_kernel<<<dim3(D_MODEL / 256, S_LEN), 256, 0, stream>>>(text, emb, ff32, D_MODEL);
  ln_kernel<<<S_LEN, 256, 0, stream>>>(ff32, mng, mnb, x32, nullptr, D_MODEL);

  for (int i = 0; i < NLAYER; i++) {
    const float* wq_i = wq + (size_t)i * D_MODEL * HD;
    const float* wk_i = wk + (size_t)i * D_MODEL * KD;
    const float* wv_i = wv + (size_t)i * D_MODEL * KD;
    const float* wo_i = wo + (size_t)i * HD * D_MODEL;
    const float* w1_i = w1 + (size_t)i * D_MODEL * FF_INNER;
    const float* w2_i = w2 + (size_t)i * FF_INNER * D_MODEL;

    // h = LN(x, blk);  hn = LN(h, attn) -> bf16
    ln_kernel<<<S_LEN, 256, 0, stream>>>(x32, bng + i * D_MODEL, bnb + i * D_MODEL,
                                         h32, nullptr, D_MODEL);
    ln_kernel<<<S_LEN, 256, 0, stream>>>(h32, ang + i * D_MODEL, anb + i * D_MODEL,
                                         nullptr, hn16, D_MODEL);

    // Q = hn @ wq  (f32 out for qk-norm)
    tc_kernel<<<tc_grid(D_MODEL, HD), 256, 0, stream>>>(wq_i, wT, D_MODEL, HD);
    gemm_bf16<2><<<gemm_grid(S_LEN, HD), 256, 0, stream>>>(hn16, wT, q32, nullptr,
                                                           nullptr, nullptr,
                                                           S_LEN, HD, D_MODEL);
    qknorm_kernel<<<(S_LEN * NH) / 8, 256, 0, stream>>>(q32, qng + i * DHEAD, q16, S_LEN * NH);

    // K = hn @ wk
    tc_kernel<<<tc_grid(D_MODEL, KD), 256, 0, stream>>>(wk_i, wT, D_MODEL, KD);
    gemm_bf16<2><<<gemm_grid(S_LEN, KD), 256, 0, stream>>>(hn16, wT, k32, nullptr,
                                                           nullptr, nullptr,
                                                           S_LEN, KD, D_MODEL);
    qknorm_kernel<<<(S_LEN * NKH) / 8, 256, 0, stream>>>(k32, kng + i * DHEAD, k16, S_LEN * NKH);

    // V = hn @ wv (straight to bf16)
    tc_kernel<<<tc_grid(D_MODEL, KD), 256, 0, stream>>>(wv_i, wT, D_MODEL, KD);
    gemm_bf16<0><<<gemm_grid(S_LEN, KD), 256, 0, stream>>>(hn16, wT, nullptr, v16,
                                                           nullptr, nullptr,
                                                           S_LEN, KD, D_MODEL);

    // attention
    attn_kernel<<<dim3(S_LEN / 16, NH), 32, 0, stream>>>(q16, k16, v16, a16);

    // h2 = attn @ wo + h  (f32), plus bf16 copy for FFN
    tc_kernel<<<tc_grid(HD, D_MODEL), 256, 0, stream>>>(wo_i, wT, HD, D_MODEL);
    gemm_bf16<2><<<gemm_grid(S_LEN, D_MODEL), 256, 0, stream>>>(a16, wT, h232, nullptr,
                                                                nullptr, h32,
                                                                S_LEN, D_MODEL, HD);
    cast_kernel<<<(S_LEN * D_MODEL) / 256, 256, 0, stream>>>(h232, h216, S_LEN * D_MODEL);

    // g = gelu(h2 @ w1 + b1) -> bf16
    tc_kernel<<<tc_grid(D_MODEL, FF_INNER), 256, 0, stream>>>(w1_i, wT, D_MODEL, FF_INNER);
    gemm_bf16<1><<<gemm_grid(S_LEN, FF_INNER), 256, 0, stream>>>(h216, wT, nullptr, g16,
                                                                 b1 + (size_t)i * FF_INNER,
                                                                 nullptr,
                                                                 S_LEN, FF_INNER, D_MODEL);

    // ff = g @ w2 + b2 + h2  (f32)
    tc_kernel<<<tc_grid(FF_INNER, D_MODEL), 256, 0, stream>>>(w2_i, wT, FF_INNER, D_MODEL);
    gemm_bf16<2><<<gemm_grid(S_LEN, D_MODEL), 256, 0, stream>>>(g16, wT, ff32, nullptr,
                                                                b2 + (size_t)i * D_MODEL,
                                                                h232,
                                                                S_LEN, D_MODEL, FF_INNER);

    // x = LN(ff, model)
    ln_kernel<<<S_LEN, 256, 0, stream>>>(ff32, mng, mnb, x32, nullptr, D_MODEL);
  }

  // head: LN -> bf16, logits = xh @ head_w + head_b
  ln_kernel<<<S_LEN, 256, 0, stream>>>(x32, hng, hnb, nullptr, xh16, D_MODEL);
  tc_kernel<<<tc_grid(D_MODEL, VOCAB), 256, 0, stream>>>(hw, wT, D_MODEL, VOCAB);
  gemm_bf16<2><<<gemm_grid(S_LEN, VOCAB), 256, 0, stream>>>(xh16, wT, out, nullptr,
                                                            hb, nullptr,
                                                            S_LEN, VOCAB, D_MODEL);
}
